// SupConLoss2_10239202034247
// MI455X (gfx1250) — matrix operand order repaired and checked by reading, not compiled, not run on hardware
//
#include <hip/hip_runtime.h>
#include <stdint.h>

#define BATCH 8192
#define DIM   256
#define INV_T (1.0f / 0.07f)
#define NITER (BATCH / 64)
#define LDS_PITCH_H 264   // halves per staged B row: 256 data + 8 pad (528B = 4 banks skew/lane)

typedef __attribute__((ext_vector_type(16))) __bf16 v16bf;
typedef __attribute__((ext_vector_type(8)))  __bf16 v8bf;
typedef __attribute__((ext_vector_type(8)))  float  v8f;

// Build a 16-element bf16 WMMA fragment from two 16-byte loads (global or LDS).
__device__ __forceinline__ v16bf ld16bf(const __bf16* p0, const __bf16* p1) {
    v8bf lo = *(const v8bf*)p0;
    v8bf hi = *(const v8bf*)p1;
    return __builtin_shufflevector(lo, hi, 0,1,2,3,4,5,6,7,8,9,10,11,12,13,14,15);
}

// Butterfly reductions within each 16-lane half of a wave32.
__device__ __forceinline__ float redsum16(float v) {
    v += __shfl_xor(v, 1, 16);
    v += __shfl_xor(v, 2, 16);
    v += __shfl_xor(v, 4, 16);
    v += __shfl_xor(v, 8, 16);
    return v;
}
__device__ __forceinline__ float redmax16(float v) {
    v = fmaxf(v, __shfl_xor(v, 1, 16));
    v = fmaxf(v, __shfl_xor(v, 2, 16));
    v = fmaxf(v, __shfl_xor(v, 4, 16));
    v = fmaxf(v, __shfl_xor(v, 8, 16));
    return v;
}

// Async-stage one 64-row x 256-K bf16 B-slab (32KB contiguous in global) into LDS
// with a padded pitch. 256 threads x 8 x GLOBAL_LOAD_ASYNC_TO_LDS_B128 (ASYNCcnt).
// LDS byte address = low 32 bits of the generic pointer (ISA flat-aperture mapping).
__device__ __forceinline__ void stage_B_async(const __bf16* fb, const uint16_t* ldsBuf,
                                              int colBase, int tid) {
    const int row = tid >> 2;            // 0..63: staged row
    const int sub = (tid & 3) * 128;     // byte offset within the 512B row
    uint64_t ga = (uint64_t)(uintptr_t)fb + (uint64_t)(colBase + row) * (DIM * 2) + (uint64_t)sub;
    uint32_t la = (uint32_t)(uintptr_t)ldsBuf + (uint32_t)(row * (LDS_PITCH_H * 2) + sub);
#pragma unroll
    for (int i = 0; i < 8; ++i) {
        asm volatile("global_load_async_to_lds_b128 %0, %1, off"
                     :: "v"(la + i * 16), "v"(ga + (uint64_t)(i * 16))
                     : "memory");
    }
}

// f32 -> bf16 (round-to-nearest-even) conversion of the feature matrix.
__global__ __launch_bounds__(256) void cvt_bf16_kernel(const float* __restrict__ f,
                                                       uint16_t* __restrict__ fb, int n) {
    int i = blockIdx.x * blockDim.x + threadIdx.x;
    int stride = gridDim.x * blockDim.x;
    for (; i < n; i += stride) {
        uint32_t u = __float_as_uint(f[i]);
        uint32_t r = (u + 0x7FFFu + ((u >> 16) & 1u)) >> 16;
        fb[i] = (uint16_t)r;
    }
}

// Fused SupCon kernel: each block owns 32 rows, sweeps all 8192 columns twice
// (max pass, then sum pass). B tiles are double-buffered in LDS via async loads;
// sim tiles recomputed with bf16 WMMA (f32 accumulate).
__global__ __launch_bounds__(256) void supcon_main_kernel(const uint16_t* __restrict__ fbu,
                                                          const int* __restrict__ lab,
                                                          float* __restrict__ rowloss) {
    const __bf16* fb = (const __bf16*)fbu;
    const int tid      = threadIdx.x;
    const int lane     = tid & 31;
    const int wave     = tid >> 5;
    const int rowGroup = wave >> 2;       // 0..1  (16 rows each)
    const int colGroup = wave & 3;        // 0..3  (16 cols each)
    const int blockRowBase = blockIdx.x * 32;
    const int waveRowBase  = blockRowBase + rowGroup * 16;
    const int halfSel = (lane & 16) ? 1 : 0;
    const int laneN   = lane & 15;

    __shared__ __align__(16) uint16_t ldsB[2][64 * LDS_PITCH_H];  // 2 x 33KB staging
    __shared__ float ldsMax[3][4][32];   // stat, colGroup, rowInBlock
    __shared__ float ldsM[3][32];        // finalized maxima
    __shared__ float ldsSum[9][4][32];   // 9 streaming stats

    // ---- Preload A fragments: this wave's 16 rows x full K=256 (8 k-steps) ----
    // A layout (ISA 7.12.2, 16-bit 16x32): lane<16 -> M=lane, K={0..7,16..23};
    // lane>=16 -> M=lane-16, K={8..15,24..31}.
    const int aRow = waveRowBase + laneN;
    const int k0   = halfSel ? 8 : 0;
    const __bf16* ap = fb + (size_t)aRow * DIM;
    v16bf afrag[8];
#pragma unroll
    for (int s = 0; s < 8; ++s)
        afrag[s] = ld16bf(ap + s * 32 + k0, ap + s * 32 + k0 + 16);

    int rlab[8];
#pragma unroll
    for (int r = 0; r < 8; ++r)
        rlab[r] = lab[waveRowBase + halfSel * 8 + r];

    // B fragment base in LDS (per wave): lane<16 -> N=lane, K={0..7,16..23};
    // lane>=16 -> N=lane-16, K={8..15,24..31} (matches A layout).
    const int bRowOff = (colGroup * 16 + laneN) * LDS_PITCH_H + (halfSel ? 8 : 0);

    // ================= Sweep 1: per-row maxima =================
    float mAll[8], mS[8], mD[8];
#pragma unroll
    for (int r = 0; r < 8; ++r) { mAll[r] = -1e30f; mS[r] = -1e30f; mD[r] = -1e30f; }

    stage_B_async(fb, ldsB[0], 0, tid);
    for (int ct = 0; ct < NITER; ++ct) {
        if (ct + 1 < NITER) {
            stage_B_async(fb, ldsB[(ct + 1) & 1], (ct + 1) * 64, tid);
            asm volatile("s_wait_asynccnt 0x8" ::: "memory");  // current slab done
        } else {
            asm volatile("s_wait_asynccnt 0x0" ::: "memory");
        }
        __syncthreads();

        const __bf16* bbase = (const __bf16*)(&ldsB[ct & 1][0]) + bRowOff;
        v8f c = {};
#pragma unroll
        for (int s = 0; s < 8; ++s) {
            v16bf bfrag = ld16bf(bbase + s * 32, bbase + s * 32 + 16);
            c = __builtin_amdgcn_wmma_f32_16x16x32_bf16(false, afrag[s], false, bfrag,
                                                        (short)0, c, false, false);
        }
        const int colBase = ct * 64 + colGroup * 16;
        const int gcol    = colBase + laneN;
        const int cl      = lab[gcol];
#pragma unroll
        for (int r = 0; r < 8; ++r) {
            float sv   = c[r];
            int   grow = waveRowBase + halfSel * 8 + r;
            bool  same = (rlab[r] == cl) && (grow != gcol);
            bool  diff = (rlab[r] != cl);
            mAll[r] = fmaxf(mAll[r], sv);
            if (same) mS[r] = fmaxf(mS[r], sv);
            if (diff) mD[r] = fmaxf(mD[r], sv);
        }
        __syncthreads();
    }
#pragma unroll
    for (int r = 0; r < 8; ++r) {
        mAll[r] = redmax16(mAll[r]); mS[r] = redmax16(mS[r]); mD[r] = redmax16(mD[r]);
    }
    if (laneN == 0) {
#pragma unroll
        for (int r = 0; r < 8; ++r) {
            int rowLoc = rowGroup * 16 + halfSel * 8 + r;
            ldsMax[0][colGroup][rowLoc] = mAll[r];
            ldsMax[1][colGroup][rowLoc] = mS[r];
            ldsMax[2][colGroup][rowLoc] = mD[r];
        }
    }
    __syncthreads();
    if (tid < 32) {
#pragma unroll
        for (int s = 0; s < 3; ++s) {
            float m = ldsMax[s][0][tid];
            m = fmaxf(m, ldsMax[s][1][tid]);
            m = fmaxf(m, ldsMax[s][2][tid]);
            m = fmaxf(m, ldsMax[s][3][tid]);
            ldsM[s][tid] = m;
        }
    }
    __syncthreads();

    float m1r[8], mxsr[8], mxdr[8];
#pragma unroll
    for (int r = 0; r < 8; ++r) {
        int rowLoc = rowGroup * 16 + halfSel * 8 + r;
        m1r[r]  = ldsM[0][rowLoc];
        mxsr[r] = ldsM[1][rowLoc];
        mxdr[r] = ldsM[2][rowLoc];
    }

    // ================= Sweep 2: streaming sufficient statistics =================
    // With dm = s - mx, z = e^dm, q = e^{dm/T}  (2 transcendentals per element):
    //   same: Qs=Σq, Zs=Σz, ZQs=Σz*q, DM=Σdm, DMZ=Σdm*z, Cnt
    //   diff: Qd=Σq, Zd=Σz, ZQd=Σz*q
    // Row constants e^{(mx-m1)/T} are hoisted to the finalize step.
    float aQs[8] = {}, aQd[8] = {}, aZs[8] = {}, aZd[8] = {}, aZQs[8] = {},
          aZQd[8] = {}, aDM[8] = {}, aDMZ[8] = {}, aCnt[8] = {};

    stage_B_async(fb, ldsB[0], 0, tid);
    for (int ct = 0; ct < NITER; ++ct) {
        if (ct + 1 < NITER) {
            stage_B_async(fb, ldsB[(ct + 1) & 1], (ct + 1) * 64, tid);
            asm volatile("s_wait_asynccnt 0x8" ::: "memory");
        } else {
            asm volatile("s_wait_asynccnt 0x0" ::: "memory");
        }
        __syncthreads();

        const __bf16* bbase = (const __bf16*)(&ldsB[ct & 1][0]) + bRowOff;
        v8f c = {};
#pragma unroll
        for (int s = 0; s < 8; ++s) {
            v16bf bfrag = ld16bf(bbase + s * 32, bbase + s * 32 + 16);
            c = __builtin_amdgcn_wmma_f32_16x16x32_bf16(false, afrag[s], false, bfrag,
                                                        (short)0, c, false, false);
        }
        const int colBase = ct * 64 + colGroup * 16;
        const int gcol    = colBase + laneN;
        const int cl      = lab[gcol];
#pragma unroll
        for (int r = 0; r < 8; ++r) {
            float sv   = c[r];
            int   grow = waveRowBase + halfSel * 8 + r;
            bool  same = (rlab[r] == cl) && (grow != gcol);
            bool  diff = (rlab[r] != cl);
            if (same) {
                float dm = sv - mxsr[r];
                float z  = __expf(dm);
                float q  = __expf(dm * INV_T);
                aZs[r]  += z;
                aQs[r]  += q;
                aZQs[r] += z * q;
                aDM[r]  += dm;
                aDMZ[r] += dm * z;
                aCnt[r] += 1.0f;
            }
            if (diff) {
                float dm = sv - mxdr[r];
                float z  = __expf(dm);
                float q  = __expf(dm * INV_T);
                aZd[r]  += z;
                aQd[r]  += q;
                aZQd[r] += z * q;
            }
        }
        __syncthreads();
    }
#pragma unroll
    for (int r = 0; r < 8; ++r) {
        aQs[r] = redsum16(aQs[r]);   aQd[r] = redsum16(aQd[r]);
        aZs[r] = redsum16(aZs[r]);   aZd[r] = redsum16(aZd[r]);
        aZQs[r] = redsum16(aZQs[r]); aZQd[r] = redsum16(aZQd[r]);
        aDM[r] = redsum16(aDM[r]);   aDMZ[r] = redsum16(aDMZ[r]);
        aCnt[r] = redsum16(aCnt[r]);
    }
    if (laneN == 0) {
#pragma unroll
        for (int r = 0; r < 8; ++r) {
            int rowLoc = rowGroup * 16 + halfSel * 8 + r;
            ldsSum[0][colGroup][rowLoc] = aQs[r];
            ldsSum[1][colGroup][rowLoc] = aQd[r];
            ldsSum[2][colGroup][rowLoc] = aZs[r];
            ldsSum[3][colGroup][rowLoc] = aZd[r];
            ldsSum[4][colGroup][rowLoc] = aZQs[r];
            ldsSum[5][colGroup][rowLoc] = aZQd[r];
            ldsSum[6][colGroup][rowLoc] = aDM[r];
            ldsSum[7][colGroup][rowLoc] = aDMZ[r];
            ldsSum[8][colGroup][rowLoc] = aCnt[r];
        }
    }
    __syncthreads();

    // ---- finalize per-row loss (32 rows per block) ----
    if (tid < 32) {
        float st[9];
#pragma unroll
        for (int s = 0; s < 9; ++s)
            st[s] = ldsSum[s][0][tid] + ldsSum[s][1][tid] +
                    ldsSum[s][2][tid] + ldsSum[s][3][tid];
        float Qs = st[0], Qd = st[1], Zs = st[2], Zd = st[3], ZQs = st[4],
              ZQd = st[5], DM = st[6], DMZ = st[7], Cnt = st[8];
        float m1 = ldsM[0][tid], mxs = ldsM[1][tid], mxd = ldsM[2][tid];

        float cs = __expf((mxs - m1) * INV_T);
        float cd = __expf((mxd - m1) * INV_T);
        // S = Σ e^{logits} w,  w = (1 - p_same) on same / (1 + p_diff) on diff
        float S  = cs * (Qs - ZQs / Zs) + cd * (Qd + ZQd / Zd);
        float SL  = DM + Cnt * (mxs - m1);      // Σ_same (s - m1)
        float SLE = DMZ + (mxs - m1) * Zs;      // Σ_same (s - m1) e^{s-mxs}
        float A = (SL - SLE / Zs) * INV_T;      // Σ logits * w_same
        float W = Cnt - 1.0f;                   // Σ w_same (Σ p_same == 1)
        float loss = (Cnt > 0.0f) ? -(A - __logf(S) * W) / Cnt : 0.0f;
        rowloss[blockRowBase + tid] = loss;
    }
}

// Final mean over the 8192 per-row losses.
__global__ __launch_bounds__(256) void final_reduce_kernel(const float* __restrict__ rowloss,
                                                           float* __restrict__ out) {
    __shared__ float ls[8];
    float v = 0.0f;
    for (int i = threadIdx.x; i < BATCH; i += 256) v += rowloss[i];
    for (int m = 1; m < 32; m <<= 1) v += __shfl_xor(v, m, 32);
    if ((threadIdx.x & 31) == 0) ls[threadIdx.x >> 5] = v;
    __syncthreads();
    if (threadIdx.x == 0) {
        float t = 0.0f;
#pragma unroll
        for (int w = 0; w < 8; ++w) t += ls[w];
        out[0] = t / (float)BATCH;
    }
}

extern "C" void kernel_launch(void* const* d_in, const int* in_sizes, int n_in,
                              void* d_out, int out_size, void* d_ws, size_t ws_size,
                              hipStream_t stream) {
    const float* feats  = (const float*)d_in[0];
    const int*   labels = (const int*)d_in[1];
    float*       out    = (float*)d_out;

    uint16_t* fb      = (uint16_t*)d_ws;                                    // 4 MB bf16 features
    float*    rowloss = (float*)((char*)d_ws + (size_t)BATCH * DIM * 2);    // 32 KB

    cvt_bf16_kernel<<<1024, 256, 0, stream>>>(feats, fb, BATCH * DIM);
    supcon_main_kernel<<<BATCH / 32, 256, 0, stream>>>(fb, labels, rowloss);
    final_reduce_kernel<<<1, 256, 0, stream>>>(rowloss, out);
}
